// Attention_24670292148859
// MI455X (gfx1250) — matrix operand order repaired and checked by reading, not compiled, not run on hardware
//
#include <hip/hip_runtime.h>
#include <hip/hip_bf16.h>

typedef __attribute__((ext_vector_type(16))) _Float16 v16h;
typedef __attribute__((ext_vector_type(8)))  _Float16 v8h;
typedef __attribute__((ext_vector_type(8)))  float    v8f;

#define ATT_B 2
#define ATT_N 2048
#define ATT_C 1024
#define ATT_H 16
#define ATT_D 64
#define ATT_BN (ATT_B * ATT_N)          // 4096

static __device__ __forceinline__ v8f wmma_f16(v16h a, v16h b, v8f c) {
    return __builtin_amdgcn_wmma_f32_16x16x32_f16(false, a, false, b, (short)0, c, false, false);
}

// Build a v16h from two contiguous 16B (8-half) chunks (16B-aligned each).
union V16U { v16h v; v8h h[2]; };
static __device__ __forceinline__ v16h ld2x8(const _Float16* p0, const _Float16* p1) {
    V16U u;
    u.h[0] = *(const v8h*)p0;
    u.h[1] = *(const v8h*)p1;
    return u.v;
}

// CDNA5 async global->LDS copy (ASYNCcnt-tracked), 16 bytes per lane.
static __device__ __forceinline__ void async_load_b128(unsigned ldsOff, unsigned long long gaddr) {
    asm volatile("global_load_async_to_lds_b128 %0, %1, off"
                 :: "v"(ldsOff), "v"(gaddr) : "memory");
}
static __device__ __forceinline__ void wait_asynccnt0() {
    asm volatile("s_wait_asynccnt 0x0" ::: "memory");
}

// ---------------------------------------------------------------------------
// Kernel 1: fused QKV projection.
//   X (4096 x 1024, f32) @ [Wq | Wkv] + bias -> f16:
//     Q (pre-scaled by D^-0.5) and K as [B][H][N][D]; V as [B][H][D][N]
//     (V d-major so attention's V tiles async-copy to LDS already transposed).
// Block 128 (4 waves): 64 rows x 64 cols per block. Grid (48, 64).
// ---------------------------------------------------------------------------
__global__ void qkv_proj_kernel(const float* __restrict__ x,
                                const float* __restrict__ Wq,
                                const float* __restrict__ bq,
                                const float* __restrict__ Wkv,
                                const float* __restrict__ bkv,
                                _Float16* __restrict__ qf,
                                _Float16* __restrict__ kf,
                                _Float16* __restrict__ vf) {
    __shared__ __align__(32) _Float16 As[64][40];   // [row][k], rows 80B
    __shared__ __align__(32) _Float16 Wt[64][48];   // [col][k], rows 96B (32B aligned)

    const int tid  = threadIdx.x;
    const int wave = tid >> 5;
    const int lane = tid & 31;
    const int half = lane >> 4;
    const int ln   = lane & 15;

    const int nbase = blockIdx.x * 64;   // 0..3071
    const int mbase = blockIdx.y * 64;   // 0..4095
    const int seg   = nbase >> 10;       // 0=Q, 1=K, 2=V  (block-uniform)

    const float* wsrc; int wld, woff;
    if (seg == 0) { wsrc = Wq;  wld = 1024; woff = 0;    }
    else          { wsrc = Wkv; wld = 2048; woff = 1024; }

    v8f acc[4];
    for (int s = 0; s < 4; ++s)
        for (int v2 = 0; v2 < 8; ++v2) acc[s][v2] = 0.0f;

    for (int kk = 0; kk < ATT_C; kk += 32) {
        {   // stage A tile (f32 -> f16), contiguous
            int r = tid >> 1, c0 = (tid & 1) * 16;
            const float* src = x + (size_t)(mbase + r) * ATT_C + kk + c0;
            #pragma unroll
            for (int i = 0; i < 16; ++i) As[r][c0 + i] = (_Float16)src[i];
        }
        {   // stage W tile transposed: coalesced global read, LDS scatter
            int r = tid >> 2, c0 = (tid & 3) * 16;
            const float* src = wsrc + (size_t)(kk + r) * wld + (nbase - woff) + c0;
            #pragma unroll
            for (int i = 0; i < 16; ++i) Wt[c0 + i][r] = (_Float16)src[i];
        }
        __syncthreads();

        // Issue ALL fragment loads first (one dscnt wait), then 4 WMMAs.
        const int row = wave * 16 + ln;
        v16h afrag = ld2x8(&As[row][half * 16], &As[row][half * 16 + 8]);
        v16h bfrag[4];
        #pragma unroll
        for (int s = 0; s < 4; ++s)
            bfrag[s] = *(const v16h*)&Wt[s * 16 + ln][half * 16];
        #pragma unroll
        for (int s = 0; s < 4; ++s)
            acc[s] = wmma_f16(afrag, bfrag[s], acc[s]);
        __syncthreads();
    }

    // Epilogue: routing is block-uniform (scalar branch, no lane divergence).
    const int r0 = mbase + wave * 16 + half * 8;       // first of this lane's 8 rows
    if (seg == 0) {            // Q: (val + bq) * D^-0.5 -> [B][H][N][D]
        #pragma unroll
        for (int s = 0; s < 4; ++s) {
            int c = (nbase + s * 16 + ln);             // 0..1023
            int head = c >> 6, d = c & 63;
            float bb = bq[c];
            #pragma unroll
            for (int v2 = 0; v2 < 8; ++v2) {
                int r = r0 + v2, b = r >> 11, nn = r & (ATT_N - 1);
                qf[(((size_t)(b * ATT_H + head)) * ATT_N + nn) * ATT_D + d] =
                    (_Float16)((acc[s][v2] + bb) * 0.125f);
            }
        }
    } else if (seg == 1) {     // K: val + bkv[c] -> [B][H][N][D]
        #pragma unroll
        for (int s = 0; s < 4; ++s) {
            int c = (nbase - 1024) + s * 16 + ln;      // 0..1023
            int head = c >> 6, d = c & 63;
            float bb = bkv[c];
            #pragma unroll
            for (int v2 = 0; v2 < 8; ++v2) {
                int r = r0 + v2, b = r >> 11, nn = r & (ATT_N - 1);
                kf[(((size_t)(b * ATT_H + head)) * ATT_N + nn) * ATT_D + d] =
                    (_Float16)(acc[s][v2] + bb);
            }
        }
    } else {                   // V: val + bkv[1024+c] -> [B][H][D][N] (d-major)
        #pragma unroll
        for (int s = 0; s < 4; ++s) {
            int c = (nbase - 2048) + s * 16 + ln;      // 0..1023
            int head = c >> 6, d = c & 63;
            float bb = bkv[1024 + c];
            #pragma unroll
            for (int v2 = 0; v2 < 8; ++v2) {
                int r = r0 + v2, b = r >> 11, nn = r & (ATT_N - 1);
                vf[(((size_t)(b * ATT_H + head)) * ATT_D + d) * ATT_N + nn] =
                    (_Float16)(acc[s][v2] + bb);
            }
        }
    }
}

// ---------------------------------------------------------------------------
// Kernel 2: flash attention. Bias enters as the WMMA C operand (Q pre-scaled).
// V tiles stream into LDS with async global->LDS copies (overlapped with the
// S = Q K^T WMMAs), already transposed thanks to V's d-major layout.
// Block 128 (wave w owns 16 query rows). Grid (N/64 = 32, B*H = 32).
// ---------------------------------------------------------------------------
__global__ void flash_attn_kernel(const _Float16* __restrict__ qf,
                                  const _Float16* __restrict__ kf,
                                  const _Float16* __restrict__ vf,
                                  const float* __restrict__ bias,
                                  _Float16* __restrict__ af) {
    __shared__ __align__(32) _Float16 Vt[64][48];        // [d][key], rows 96B
    __shared__ __align__(32) _Float16 Pbuf[4][16][32];   // per-wave P transpose

    const int tid  = threadIdx.x;
    const int wave = tid >> 5;
    const int lane = tid & 31;
    const int half = lane >> 4;
    const int ln   = lane & 15;

    const int bh    = blockIdx.y;
    const int qbase = blockIdx.x * 64 + wave * 16;
    const size_t headBase = (size_t)bh * ATT_N * ATT_D;
    const size_t biasBase = (size_t)bh * ATT_N * ATT_N;

    // Resident Q fragments for D-chunks [0,32) and [32,64).
    const _Float16* qrow = qf + headBase + (size_t)(qbase + ln) * ATT_D;
    v16h q0 = ld2x8(qrow + half * 16,      qrow + half * 16 + 8);
    v16h q1 = ld2x8(qrow + 32 + half * 16, qrow + 40 + half * 16);

    v8f o[4];
    for (int t = 0; t < 4; ++t)
        for (int v2 = 0; v2 < 8; ++v2) o[t][v2] = 0.0f;
    float m_i[8], l_i[8];
    for (int v2 = 0; v2 < 8; ++v2) { m_i[v2] = -1e30f; l_i[v2] = 0.0f; }

    // Async V-tile addressing: thread covers d = tid>>1, 16-key half c = tid&1.
    const int vd = tid >> 1, vc = tid & 1;
    const _Float16* vrow = vf + headBase + (size_t)vd * ATT_N + vc * 16;
    const unsigned vtOff0 = (unsigned)(unsigned long long)(uintptr_t)&Vt[vd][vc * 16];
    const unsigned vtOff1 = vtOff0 + 16;

    for (int k0 = 0; k0 < ATT_N; k0 += 32) {
        // Kick off async V staging (memory -> LDS, already transposed).
        async_load_b128(vtOff0, (unsigned long long)(uintptr_t)(vrow + k0));
        async_load_b128(vtOff1, (unsigned long long)(uintptr_t)(vrow + k0 + 8));

        // S = bias + Qhat @ K^T (two 16-key sub-tiles, K-dim 64 = 2 chunks)
        v8f s[2];
        #pragma unroll
        for (int sub = 0; sub < 2; ++sub) {
            v8f c;
            const float* bp = bias + biasBase + (size_t)(qbase + half * 8) * ATT_N
                              + k0 + sub * 16 + ln;
            #pragma unroll
            for (int v2 = 0; v2 < 8; ++v2) c[v2] = bp[(size_t)v2 * ATT_N];

            const _Float16* krow = kf + headBase + (size_t)(k0 + sub * 16 + ln) * ATT_D;
            v16h kb0 = *(const v16h*)(krow + half * 16);
            v16h kb1 = *(const v16h*)(krow + 32 + half * 16);
            c = wmma_f16(q0, kb0, c);
            c = wmma_f16(q1, kb1, c);
            s[sub] = c;
        }

        // Online softmax (rows replicated across the 16 lanes of a half-wave).
        #pragma unroll
        for (int v2 = 0; v2 < 8; ++v2) {
            float t = fmaxf(s[0][v2], s[1][v2]);
            for (int off = 1; off < 16; off <<= 1) t = fmaxf(t, __shfl_xor(t, off, 16));
            float nm    = fmaxf(m_i[v2], t);
            float alpha = __expf(m_i[v2] - nm);
            m_i[v2] = nm;
            float p0 = __expf(s[0][v2] - nm);
            float p1 = __expf(s[1][v2] - nm);
            s[0][v2] = p0; s[1][v2] = p1;
            float rs = p0 + p1;
            for (int off = 1; off < 16; off <<= 1) rs += __shfl_xor(rs, off, 16);
            l_i[v2] = l_i[v2] * alpha + rs;
            #pragma unroll
            for (int t4 = 0; t4 < 4; ++t4) o[t4][v2] *= alpha;
        }

        // P (C/D layout) -> per-wave LDS buffer.
        #pragma unroll
        for (int v2 = 0; v2 < 8; ++v2) {
            int m = v2 + half * 8;
            Pbuf[wave][m][ln]      = (_Float16)s[0][v2];
            Pbuf[wave][m][16 + ln] = (_Float16)s[1][v2];
        }

        wait_asynccnt0();        // V tile landed in LDS
        __syncthreads();

        // Issue all LDS fragment loads up front, then 4 back-to-back WMMAs.
        v16h pa = ld2x8(&Pbuf[wave][ln][half * 16], &Pbuf[wave][ln][half * 16 + 8]);
        v16h vb[4];
        #pragma unroll
        for (int t4 = 0; t4 < 4; ++t4)
            vb[t4] = *(const v16h*)&Vt[t4 * 16 + ln][half * 16];
        #pragma unroll
        for (int t4 = 0; t4 < 4; ++t4)
            o[t4] = wmma_f16(pa, vb[t4], o[t4]);
        __syncthreads();

        // Prefetch next bias tile toward L2 while WMMAs drain.
        if (k0 + 32 < ATT_N)
            __builtin_prefetch(bias + biasBase + (size_t)(qbase + half * 8) * ATT_N
                               + k0 + 32 + ln, 0, 1);
    }

    // Normalize and store f16 in (B, N, C) layout for the output projection.
    const int b = bh >> 4, head = bh & 15;
    #pragma unroll
    for (int v2 = 0; v2 < 8; ++v2) {
        float inv = 1.0f / l_i[v2];
        int row = qbase + v2 + half * 8;
        _Float16* dst = af + ((size_t)(b * ATT_N + row)) * ATT_C + head * ATT_D;
        #pragma unroll
        for (int t4 = 0; t4 < 4; ++t4)
            dst[t4 * 16 + ln] = (_Float16)(o[t4][v2] * inv);
    }
}

// ---------------------------------------------------------------------------
// Kernel 3: output projection.  attn(f16, 4096x1024) @ Wo + bo -> out (f32).
// Grid (16, 64), block 128.
// ---------------------------------------------------------------------------
__global__ void out_proj_kernel(const _Float16* __restrict__ af,
                                const float* __restrict__ Wo,
                                const float* __restrict__ bo,
                                float* __restrict__ out) {
    __shared__ __align__(32) _Float16 As[64][40];
    __shared__ __align__(32) _Float16 Wt[64][48];

    const int tid  = threadIdx.x;
    const int wave = tid >> 5;
    const int lane = tid & 31;
    const int half = lane >> 4;
    const int ln   = lane & 15;
    const int nbase = blockIdx.x * 64;
    const int mbase = blockIdx.y * 64;

    v8f acc[4];
    for (int s = 0; s < 4; ++s)
        for (int v2 = 0; v2 < 8; ++v2) acc[s][v2] = 0.0f;

    for (int kk = 0; kk < ATT_C; kk += 32) {
        {   // stage A (already f16): contiguous copy
            int r = tid >> 1, c0 = (tid & 1) * 16;
            const _Float16* src = af + (size_t)(mbase + r) * ATT_C + kk + c0;
            #pragma unroll
            for (int i = 0; i < 16; ++i) As[r][c0 + i] = src[i];
        }
        {   // stage Wo transposed
            int r = tid >> 2, c0 = (tid & 3) * 16;
            const float* src = Wo + (size_t)(kk + r) * ATT_C + nbase + c0;
            #pragma unroll
            for (int i = 0; i < 16; ++i) Wt[c0 + i][r] = (_Float16)src[i];
        }
        __syncthreads();

        const int row = wave * 16 + ln;
        v16h afrag = ld2x8(&As[row][half * 16], &As[row][half * 16 + 8]);
        v16h bfrag[4];
        #pragma unroll
        for (int s = 0; s < 4; ++s)
            bfrag[s] = *(const v16h*)&Wt[s * 16 + ln][half * 16];
        #pragma unroll
        for (int s = 0; s < 4; ++s)
            acc[s] = wmma_f16(afrag, bfrag[s], acc[s]);
        __syncthreads();
    }

    #pragma unroll
    for (int s = 0; s < 4; ++s) {
        int cg = nbase + s * 16 + ln;
        float bb = bo[cg];
        #pragma unroll
        for (int v2 = 0; v2 < 8; ++v2) {
            int r = mbase + wave * 16 + v2 + half * 8;
            out[(size_t)r * ATT_C + cg] = acc[s][v2] + bb;
        }
    }
}

// ---------------------------------------------------------------------------
extern "C" void kernel_launch(void* const* d_in, const int* in_sizes, int n_in,
                              void* d_out, int out_size, void* d_ws, size_t ws_size,
                              hipStream_t stream) {
    const float* x    = (const float*)d_in[0];
    const float* bias = (const float*)d_in[1];
    const float* Wq   = (const float*)d_in[2];
    const float* bq   = (const float*)d_in[3];
    const float* Wkv  = (const float*)d_in[4];
    const float* bkv  = (const float*)d_in[5];
    const float* Wo   = (const float*)d_in[6];
    const float* bo   = (const float*)d_in[7];
    float* out = (float*)d_out;

    // Workspace carve: Q,K,V,attn-out in f16 -> 4 * 8 MB = 32 MB.
    const size_t elems = (size_t)ATT_BN * ATT_C;
    _Float16* qf = (_Float16*)d_ws;
    _Float16* kf = qf + elems;
    _Float16* vf = kf + elems;
    _Float16* af = vf + elems;

    qkv_proj_kernel<<<dim3(48, 64), 128, 0, stream>>>(x, Wq, bq, Wkv, bkv, qf, kf, vf);
    flash_attn_kernel<<<dim3(ATT_N / 64, ATT_B * ATT_H), 128, 0, stream>>>(qf, kf, vf, bias, af);
    out_proj_kernel<<<dim3(ATT_C / 64, ATT_BN / 64), 128, 0, stream>>>(af, Wo, bo, out);
}
